// SplendorNNet_80925773791310
// MI455X (gfx1250) — matrix operand order repaired and checked by reading, not compiled, not run on hardware
//
#include <hip/hip_runtime.h>
#include <hip/hip_bf16.h>

typedef __attribute__((ext_vector_type(16))) __bf16 v16bf;
typedef __attribute__((ext_vector_type(8)))  float  v8f;
typedef __attribute__((ext_vector_type(4)))  float  v4f;

#define NWAVES 8
#define BN_INV 0.9999950000374997f

struct Params {
    const float* input;
    const unsigned char* valid;
    const __bf16* w[16];
    const float* bias[16];
    const float* g[8];
    const float* be[8];
    float* out_pi;
    float* out_v;
};

// ---------------------------------------------------------------------------
// Weight pre-pack: fp32 row-major (K x N) -> bf16 WMMA-B tiles.
// Tile (kt, nt) is 512 bf16: lane l holds N = nt*16 + (l & 15),
// K = kt*32 + (e >> 3)*16 + (l >> 4)*8 + (e & 7) for e = 0..15  (ISA B-matrix 32x16 layout).
// ---------------------------------------------------------------------------
__global__ void pack_w(const float* __restrict__ w, __bf16* __restrict__ dst,
                       int K, int N, int KT, int NT) {
    int total = KT * NT * 512;
    for (int i = blockIdx.x * blockDim.x + threadIdx.x; i < total;
         i += gridDim.x * blockDim.x) {
        int e    = i & 15;
        int ln   = (i >> 4) & 31;
        int tile = i >> 9;
        int nt = tile % NT, kt = tile / NT;
        int k = kt * 32 + (e >> 3) * 16 + (ln >> 4) * 8 + (e & 7);
        int n = nt * 16 + (ln & 15);
        float v = (k < K && n < N) ? w[(size_t)k * N + n] : 0.0f;
        dst[i] = (__bf16)v;
    }
}

// ---------------------------------------------------------------------------
// Generic tiled GEMM: A from LDS f32 canonical buffer (ds_load_b128 x4 per
// tile, converted to bf16), B from packed global bf16 tiles (b128 x2 per
// lane), C accumulated via v_wmma_f32_16x16x32_bf16.
// Epilogue: +bias, optional BN (BN==1: per-M-tile channel = mt, since stage-1
// rows are pos-major; BN==2: scalar channel), optional relu -> LDS f32.
// ---------------------------------------------------------------------------
template <int BN>
__device__ __forceinline__ void gemm(const float* __restrict__ aLds, int lda,
                                     int kofs, int KT,
                                     float* __restrict__ oLds, int ldo, int nofs,
                                     const __bf16* __restrict__ pw, int NT, int MT,
                                     int Nreal,
                                     const float* __restrict__ bias,
                                     const float* __restrict__ gp,
                                     const float* __restrict__ bep,
                                     bool relu, int wave, int lane) {
    const int l16 = lane & 15;
    const int h   = lane >> 4;

    const int total = MT * NT;
    for (int t = wave; t < total; t += NWAVES) {
        int mt = t / NT, nt = t - mt * NT;
        int row0 = mt * 16;

        float sgv = 0.f, sbv = 0.f;
        if constexpr (BN == 1) { sgv = gp[mt] * BN_INV; sbv = bep[mt]; }
        if constexpr (BN == 2) { sgv = gp[0]  * BN_INV; sbv = bep[0]; }

        v8f c = {0.f, 0.f, 0.f, 0.f, 0.f, 0.f, 0.f, 0.f};
        for (int kt = 0; kt < KT; ++kt) {
            // A tile (16x32 bf16): lane l16 = M, half h selects K sub-blocks.
            // 16B-aligned: lda and offsets are multiples of 4 floats.
            const v4f* pa = (const v4f*)(aLds + (row0 + l16) * lda + kofs +
                                         kt * 32 + h * 8);
            v4f a0 = pa[0], a1 = pa[1];   // K = h*8 + 0..7
            v4f a2 = pa[4], a3 = pa[5];   // K = 16 + h*8 + 0..7
            v16bf a;
#pragma unroll
            for (int i = 0; i < 4; ++i) {
                a[i]      = (__bf16)a0[i];
                a[4 + i]  = (__bf16)a1[i];
                a[8 + i]  = (__bf16)a2[i];
                a[12 + i] = (__bf16)a3[i];
            }
            // B tile: contiguous 32B per lane from packed weights.
            const __bf16* pb = pw + ((size_t)(kt * NT + nt) << 9) + (lane << 4);
            v16bf b = *(const v16bf*)pb;
            if (kt + 1 < KT)   // prefetch next K tile (global_prefetch_b8)
                __builtin_prefetch(pb + ((size_t)NT << 9), 0, 3);
            c = __builtin_amdgcn_wmma_f32_16x16x32_bf16(
                    false, a, false, b, (short)0, c, false, false);
        }
        // Epilogue: C/D layout -> lane l16 = N, h*8+r = M.
        int nloc = nt * 16 + l16;
        if (nloc < Nreal) {
            float bv = bias[nloc];
            int ncol = nofs + nloc;
#pragma unroll
            for (int r = 0; r < 8; ++r) {
                int row = row0 + h * 8 + r;
                float v = c[r] + bv;
                if constexpr (BN != 0) v = v * sgv + sbv;
                if (relu) v = fmaxf(v, 0.f);
                oLds[row * ldo + ncol] = v;
            }
        }
    }
}

// _dpg group pooling: out[:,g] = max of 8, out[:,8+g] = mean of 8.
__device__ __forceinline__ void dpg_pool(const float* __restrict__ in, int ldin,
                                         float* __restrict__ out, int ldout,
                                         int rows, int tid) {
    for (int i = tid; i < rows * 8; i += 256) {
        int row = i >> 3, g = i & 7;
        const float* pr = in + row * ldin + g * 8;
        float mx = pr[0], sm = pr[0];
#pragma unroll
        for (int j = 1; j < 8; ++j) { mx = fmaxf(mx, pr[j]); sm += pr[j]; }
        out[row * ldout + g]     = mx;
        out[row * ldout + 8 + g] = sm * 0.125f;
    }
}

// ---------------------------------------------------------------------------
// Main fused network kernel: one block = 16 samples.
// Stage-1 activation layout is POS-MAJOR: row = pos*16 + sample (112 rows),
// so each 16-row WMMA tile has a uniform BN channel (pos == mt).
// ---------------------------------------------------------------------------
__global__ __launch_bounds__(256) void splendor_main(Params p) {
    extern __shared__ float smem[];
    float* bufA = smem;            // 14336 floats
    float* bufB = smem + 14336;    // 14336 floats
    float* red  = smem + 28672;    // 32 floats

    const int tid  = threadIdx.x;
    const int lane = tid & 31;
    const int wave = tid >> 5;
    const long long blk = blockIdx.x;

    // Phase 0: load + transpose input -> bufA[112][128] (cols 0..55, pad to 64).
    {
        const float* in = p.input + blk * 16 * 392;
        for (int i = tid; i < 112 * 64; i += 256) {
            int row = i >> 6, k = i & 63;
            int pos = row >> 4, s = row & 15;
            bufA[row * 128 + k] = (k < 56) ? in[s * 392 + k * 7 + pos] : 0.f;
        }
    }
    __syncthreads();
    // L1: (112 x 56) @ W1 -> bufB[112][128], per-tile BN, relu
    gemm<1>(bufA, 128, 0, 2, bufB, 128, 0, p.w[0], 8, 7, 128,
            p.bias[0], p.g[0], p.be[0], true, wave, lane);
    __syncthreads();
    // L2
    gemm<1>(bufB, 128, 0, 4, bufA, 128, 0, p.w[1], 8, 7, 128,
            p.bias[1], p.g[1], p.be[1], true, wave, lane);
    __syncthreads();
    // dpg1: pooling cols 0..15, dense (cols 64..127 @ Wp1) -> cols 16..127
    dpg_pool(bufA, 128, bufB, 128, 112, tid);
    gemm<0>(bufA, 128, 64, 2, bufB, 128, 16, p.w[2], 7, 7, 112,
            p.bias[2], nullptr, nullptr, true, wave, lane);
    __syncthreads();
    // L3
    gemm<1>(bufB, 128, 0, 4, bufA, 128, 0, p.w[3], 8, 7, 128,
            p.bias[3], p.g[2], p.be[2], true, wave, lane);
    __syncthreads();
    // dpg2
    dpg_pool(bufA, 128, bufB, 128, 112, tid);
    gemm<0>(bufA, 128, 64, 2, bufB, 128, 16, p.w[4], 7, 7, 112,
            p.bias[4], nullptr, nullptr, true, wave, lane);
    __syncthreads();
    // L4
    gemm<1>(bufB, 128, 0, 4, bufA, 128, 0, p.w[5], 8, 7, 128,
            p.bias[5], p.g[3], p.be[3], true, wave, lane);
    __syncthreads();
    // Concat -> VEC[16][704] in bufB  (x[b,pos,col] = bufA[(pos*16+s)*128+col])
    for (int i = tid; i < 16 * 704; i += 256) {
        int s = i / 704, c = i - s * 704;
        float v;
        if (c < 64) {
            v = bufA[s * 128 + c];
            for (int pp = 1; pp < 5; ++pp)
                v = fmaxf(v, bufA[(pp * 16 + s) * 128 + c]);
        } else if (c < 128) {
            int cc = c - 64; float sm = 0.f;
            for (int pp = 0; pp < 5; ++pp) sm += bufA[(pp * 16 + s) * 128 + cc];
            v = sm * 0.2f;
        } else if (c < 256) {
            int cc = c - 128;
            v = bufA[(((5 + (cc >> 6)) * 16) + s) * 128 + (cc & 63)];
        } else {
            int cc = c - 256;
            v = bufA[(((cc >> 6) * 16) + s) * 128 + 64 + (cc & 63)];
        }
        bufB[s * 704 + c] = v;
    }
    __syncthreads();
    // L5: VEC(704) -> bufA[16][512], scalar BN, relu
    gemm<2>(bufB, 704, 0, 22, bufA, 512, 0, p.w[6], 32, 1, 512,
            p.bias[6], p.g[4], p.be[4], true, wave, lane);
    __syncthreads();
    // L6: 512 -> bufB[16][256]
    gemm<2>(bufA, 512, 0, 16, bufB, 256, 0, p.w[7], 16, 1, 256,
            p.bias[7], p.g[5], p.be[5], true, wave, lane);
    __syncthreads();
    // dpg4: bufB -> bufA[16][256]
    dpg_pool(bufB, 256, bufA, 256, 16, tid);
    gemm<0>(bufB, 256, 64, 6, bufA, 256, 16, p.w[8], 15, 1, 240,
            p.bias[8], nullptr, nullptr, true, wave, lane);
    __syncthreads();
    // L7
    gemm<2>(bufA, 256, 0, 8, bufB, 256, 0, p.w[9], 16, 1, 256,
            p.bias[9], p.g[6], p.be[6], true, wave, lane);
    __syncthreads();
    // L8
    gemm<2>(bufB, 256, 0, 8, bufA, 256, 0, p.w[10], 16, 1, 256,
            p.bias[10], p.g[7], p.be[7], true, wave, lane);
    __syncthreads();
    // dpg5: bufA -> bufB[16][256]  (final features)
    dpg_pool(bufA, 256, bufB, 256, 16, tid);
    gemm<0>(bufA, 256, 64, 6, bufB, 256, 16, p.w[11], 15, 1, 240,
            p.bias[11], nullptr, nullptr, true, wave, lane);
    __syncthreads();
    // Heads (no relu, no BN): hv / hpi in bufA
    float* hv  = bufA;
    float* hpi = bufA + 4096;
    gemm<0>(bufB, 256, 0, 8, hv,  256, 0, p.w[14], 16, 1, 256,
            p.bias[14], nullptr, nullptr, false, wave, lane);
    gemm<0>(bufB, 256, 0, 8, hpi, 256, 0, p.w[12], 16, 1, 256,
            p.bias[12], nullptr, nullptr, false, wave, lane);
    __syncthreads();
    float* PI = bufA + 8192;   // [16][96], cols 0..80 valid
    float* VV = bufA + 9728;   // [16][16], col 0 valid
    gemm<0>(hpi, 256, 0, 8, PI, 96, 0, p.w[13], 6, 1, 81,
            p.bias[13], nullptr, nullptr, false, wave, lane);
    gemm<0>(hv, 256, 0, 8, VV, 16, 0, p.w[15], 1, 1, 1,
            p.bias[15], nullptr, nullptr, false, wave, lane);
    __syncthreads();
    // Mask + log_softmax + tanh.
    const unsigned char* valid = p.valid + blk * 16 * 81;
    if (tid < 16) {
        int s = tid;
        float m = -1e30f;
        for (int a = 0; a < 81; ++a) {
            float pv = valid[s * 81 + a] ? PI[s * 96 + a] : -1.0e8f;
            m = fmaxf(m, pv);
        }
        float sum = 0.f;
        for (int a = 0; a < 81; ++a) {
            float pv = valid[s * 81 + a] ? PI[s * 96 + a] : -1.0e8f;
            sum += __expf(pv - m);
        }
        red[s]      = m;
        red[16 + s] = __logf(sum);
        p.out_v[blk * 16 + s] = tanhf(VV[s * 16]);
    }
    __syncthreads();
    for (int i = tid; i < 16 * 81; i += 256) {
        int s = i / 81, a = i - s * 81;
        float pv = valid[s * 81 + a] ? PI[s * 96 + a] : -1.0e8f;
        p.out_pi[(blk * 16 + s) * 81 + a] = pv - red[s] - red[16 + s];
    }
}

// ---------------------------------------------------------------------------
extern "C" void kernel_launch(void* const* d_in, const int* in_sizes, int n_in,
                              void* d_out, int out_size, void* d_ws, size_t ws_size,
                              hipStream_t stream) {
    (void)n_in; (void)out_size; (void)ws_size;
    const long long B = (long long)in_sizes[0] / (56 * 7);

    struct WInfo { int idx; int K; int N; };
    const WInfo wi[16] = {
        {2, 56, 128},  {6, 128, 128}, {10, 64, 112}, {12, 128, 128},
        {16, 64, 112}, {18, 128, 128},{22, 704, 512},{26, 512, 256},
        {30, 192, 240},{32, 256, 256},{36, 256, 256},{40, 192, 240},
        {42, 256, 256},{44, 256, 81}, {46, 256, 256},{48, 256, 1}};
    const int bidx[16]  = {3, 7, 11, 13, 17, 19, 23, 27, 31, 33, 37, 41, 43, 45, 47, 49};
    const int gidx[8]   = {4, 8, 14, 20, 24, 28, 34, 38};
    const int beidx[8]  = {5, 9, 15, 21, 25, 29, 35, 39};

    Params p;
    p.input = (const float*)d_in[0];
    p.valid = (const unsigned char*)d_in[1];

    __bf16* ws = (__bf16*)d_ws;
    size_t off = 0;
    for (int i = 0; i < 16; ++i) {
        int KT = (wi[i].K + 31) / 32;
        int NT = (wi[i].N + 15) / 16;
        size_t elems = (size_t)KT * NT * 512;
        int nblk = (int)((elems + 255) / 256);
        pack_w<<<nblk, 256, 0, stream>>>((const float*)d_in[wi[i].idx], ws + off,
                                         wi[i].K, wi[i].N, KT, NT);
        p.w[i] = ws + off;
        off += elems;
    }
    for (int i = 0; i < 16; ++i) p.bias[i] = (const float*)d_in[bidx[i]];
    for (int i = 0; i < 8; ++i)  { p.g[i] = (const float*)d_in[gidx[i]];
                                   p.be[i] = (const float*)d_in[beidx[i]]; }
    p.out_pi = (float*)d_out;
    p.out_v  = (float*)d_out + (size_t)B * 81;

    size_t shmem = (size_t)(14336 * 2 + 32) * sizeof(float);
    hipFuncSetAttribute(reinterpret_cast<const void*>(splendor_main),
                        hipFuncAttributeMaxDynamicSharedMemorySize, (int)shmem);
    splendor_main<<<(int)(B / 16), 256, shmem, stream>>>(p);
}
